// TGNLayerGraphAttentionEmbedding_65180423684468
// MI455X (gfx1250) — matrix operand order repaired and checked by reading, not compiled, not run on hardware
//
#include <hip/hip_runtime.h>
#include <math.h>

// ---------------- CDNA5 WMMA types ----------------
typedef __attribute__((ext_vector_type(16))) __bf16 v16bf;
typedef __attribute__((ext_vector_type(8)))  __bf16 v8bf;
typedef __attribute__((ext_vector_type(8)))  float  v8f;

// ---------------- generic gathered-A GEMM (optionally dual-W fused) -------
// out[m,f]  = act( ( sum_k A[m,k]*W [f,k] + bias [f] ) * alpha )
// out2[m,f] = act( ( sum_k A[m,k]*W2[f,k] + bias2[f] ) * alpha )   (FUSED)
// A[m,k] gathered from up to 3 f32 sources with repeating concat period.
struct Src { const float* p; long rowStride; int width; int _pad; };
struct GemmArgs {
  Src s0, s1, s2;
  const unsigned short* W;    // bf16 bits
  const unsigned short* W2;   // second weight (fused K/V) or null
  const float* bias;          // may be null
  const float* bias2;         // may be null
  float* out;
  float* out2;                // required iff W2
  long ldW, ldOut;
  int period, nGroup, transB, relu;
  int M, N, K;
  float alpha;
};

__device__ __forceinline__ unsigned short f32_to_bf16_rne(float x) {
  unsigned u = __float_as_uint(x);
  if ((u & 0x7fffffffu) > 0x7f800000u) return (unsigned short)((u >> 16) | 0x40); // quiet NaN
  unsigned r = u + 0x7fffu + ((u >> 16) & 1u);
  return (unsigned short)(r >> 16);
}

#define TBM 128
#define TBN 64
#define TBK 32

template <bool FUSED>
__global__ __launch_bounds__(256) void gemm_bf16_wmma_kernel(GemmArgs g) {
  __shared__ unsigned short Asm[2][TBM][TBK];                   // ping-pong A tiles
  __shared__ unsigned short Wsm[2][FUSED ? 2 : 1][TBN][TBK];    // [buf][which W][f][k]

  const int tid  = threadIdx.x;
  const int wv   = tid >> 5;                        // 8 waves (wave32)
  const int lane = tid & 31;
  const int bm = blockIdx.y * TBM;
  const int bn = blockIdx.x * TBN;

  // per-thread staging assignments
  const int arow = tid >> 1, acg = tid & 1;         // A: 16 contiguous k each
  const int wf = tid >> 2, wkg = tid & 3;           // W: 8 contiguous k each
  const int w01 = g.s0.width + g.s1.width;

  unsigned short aval[16];
  unsigned short wval[FUSED ? 2 : 1][8];

  // ---- gather one A sub-row (16 contiguous k) into regs, bf16 ----
  auto loadA = [&](int k0) {
    const int r = bm + arow;
    const int ks = k0 + acg * 16;
    if (r >= g.M) {
#pragma unroll
      for (int i = 0; i < 16; ++i) aval[i] = 0;
      return;
    }
    int n2 = 0, c = ks;
    if (g.nGroup > 1) { n2 = ks / g.period; c = ks - n2 * g.period; } // ONE div per 16 elems
    long ri = (long)r * g.nGroup + n2;
    const float* b0 = g.s0.p ? g.s0.p + ri * g.s0.rowStride : nullptr;
    const float* b1 = g.s1.p ? g.s1.p + ri * g.s1.rowStride : nullptr;
    const float* b2 = g.s2.p ? g.s2.p + ri * g.s2.rowStride : nullptr;
#pragma unroll
    for (int i = 0; i < 16; ++i) {
      float v = 0.0f;
      if (ks + i < g.K) {
        if (c < g.s0.width)      v = b0 ? b0[c] : 0.0f;
        else if (c < w01)        v = b1 ? b1[c - g.s0.width] : 0.0f;
        else                     v = b2 ? b2[c - w01] : 0.0f;
      }
      aval[i] = f32_to_bf16_rne(v);
      if (++c == g.period && g.nGroup > 1) {          // walk into next neighbor block
        c = 0; ++n2; ri += 1;
        b0 = g.s0.p ? g.s0.p + ri * g.s0.rowStride : nullptr;
        b1 = g.s1.p ? g.s1.p + ri * g.s1.rowStride : nullptr;
        b2 = g.s2.p ? g.s2.p + ri * g.s2.rowStride : nullptr;
      }
    }
  };

  // ---- load W sub-rows (8 contiguous k) into regs ----
  auto loadW = [&](int k0) {
    const int fg = bn + wf;
#pragma unroll
    for (int c = 0; c < 8; ++c) {
      const int k = k0 + wkg * 8 + c;
      unsigned short v = 0, v2 = 0;
      if (fg < g.N && k < g.K) {
        long idx = g.transB ? ((long)k * g.ldW + fg) : ((long)fg * g.ldW + k);
        v = g.W[idx];
        if constexpr (FUSED) v2 = g.W2[idx];
      }
      wval[0][c] = v;
      if constexpr (FUSED) wval[1][c] = v2;
    }
    if (!g.transB && fg < g.N && (k0 + TBK) < g.K)     // global_prefetch_b8
      __builtin_prefetch(&g.W[(long)fg * g.ldW + k0 + TBK], 0, 1);
  };

  auto stageLds = [&](int buf) {
#pragma unroll
    for (int i = 0; i < 16; ++i) Asm[buf][arow][acg * 16 + i] = aval[i];
#pragma unroll
    for (int c = 0; c < 8; ++c) Wsm[buf][0][wf][wkg * 8 + c] = wval[0][c];
    if constexpr (FUSED) {
#pragma unroll
      for (int c = 0; c < 8; ++c) Wsm[buf][1][wf][wkg * 8 + c] = wval[1][c];
    }
  };

  v8f acc[FUSED ? 2 : 1][4];
#pragma unroll
  for (int w = 0; w < (FUSED ? 2 : 1); ++w)
#pragma unroll
    for (int j = 0; j < 4; ++j)
#pragma unroll
      for (int i = 0; i < 8; ++i) acc[w][j][i] = 0.0f;

  const int ktiles = (g.K + TBK - 1) / TBK;
  loadA(0); loadW(0); stageLds(0);
  __syncthreads();

  for (int kt = 0; kt < ktiles; ++kt) {
    const int cur = kt & 1;
    if (kt + 1 < ktiles) { loadA((kt + 1) * TBK); loadW((kt + 1) * TBK); }

    // ---- fragments + WMMA from LDS[cur] (ISA 7.12.2 16-bit layouts) ----
    union { v16bf v; v8bf h[2]; } af, bf_;
    const int m   = wv * 16 + (lane & 15);
    const int akb = (lane < 16) ? 0 : 8;
    af.h[0] = *reinterpret_cast<const v8bf*>(&Asm[cur][m][akb]);
    af.h[1] = *reinterpret_cast<const v8bf*>(&Asm[cur][m][akb + 16]);
    const int bkb = (lane < 16) ? 0 : 8;
#pragma unroll
    for (int j = 0; j < 4; ++j) {
      const int f = j * 16 + (lane & 15);
      bf_.h[0] = *reinterpret_cast<const v8bf*>(&Wsm[cur][0][f][bkb]);
      bf_.h[1] = *reinterpret_cast<const v8bf*>(&Wsm[cur][0][f][bkb + 16]);
      acc[0][j] = __builtin_amdgcn_wmma_f32_16x16x32_bf16(
          false, af.v, false, bf_.v, (short)0, acc[0][j], false, false);
      if constexpr (FUSED) {
        bf_.h[0] = *reinterpret_cast<const v8bf*>(&Wsm[cur][1][f][bkb]);
        bf_.h[1] = *reinterpret_cast<const v8bf*>(&Wsm[cur][1][f][bkb + 16]);
        acc[1][j] = __builtin_amdgcn_wmma_f32_16x16x32_bf16(
            false, af.v, false, bf_.v, (short)0, acc[1][j], false, false);
      }
    }

    if (kt + 1 < ktiles) stageLds(cur ^ 1);
    __syncthreads();
  }

  // ---- epilogue: C/D layout (VGPR i: lanes0-15 M=i, lanes16-31 M=8+i) ----
#pragma unroll
  for (int w = 0; w < (FUSED ? 2 : 1); ++w) {
    const float* bias = (w == 0) ? g.bias : g.bias2;
    float* o = (w == 0) ? g.out : g.out2;
#pragma unroll
    for (int j = 0; j < 4; ++j) {
      const int n = bn + j * 16 + (lane & 15);
      if (n >= g.N) continue;
      const float b = bias ? bias[n] : 0.0f;
      const int mbase = bm + wv * 16 + ((lane >> 4) << 3);
#pragma unroll
      for (int i = 0; i < 8; ++i) {
        const int m = mbase + i;
        if (m >= g.M) continue;
        float v = (acc[w][j][i] + b) * g.alpha;
        if (g.relu) v = fmaxf(v, 0.0f);
        o[(long)m * g.ldOut + n] = v;
      }
    }
  }
}

// ---------------- small per-source attention (prop 0) ----------------
__global__ __launch_bounds__(256) void attn0_kernel(const float* __restrict__ qp,
                                                    const float* __restrict__ kp,
                                                    const float* __restrict__ vp,
                                                    float* __restrict__ o) {
  const int s = blockIdx.x;
  __shared__ float sc[2][10][10];
  const int t = threadIdx.x;
  if (t < 200) {
    const int h = t / 100, rem = t % 100, q = rem / 10, k = rem % 10;
    const float* qr = qp + (long)(s * 10 + q) * 272 + h * 136;
    const float* kr = kp + (long)(s * 10 + k) * 272 + h * 136;
    float d = 0.0f;
    for (int i = 0; i < 136; ++i) d += qr[i] * kr[i];
    sc[h][q][k] = d;
  }
  __syncthreads();
  if (t < 20) {
    const int h = t / 10, q = t % 10;
    float mx = -3.0e38f;
    for (int k = 0; k < 10; ++k) mx = fmaxf(mx, sc[h][q][k]);
    float e[10], sum = 0.0f;
    for (int k = 0; k < 10; ++k) { e[k] = expf(sc[h][q][k] - mx); sum += e[k]; }
    const float inv = 1.0f / sum;
    for (int k = 0; k < 10; ++k) sc[h][q][k] = e[k] * inv;
  }
  __syncthreads();
  for (int idx = t; idx < 2720; idx += 256) {
    const int q = idx / 272, c = idx % 272;
    const int h = c / 136;
    float a = 0.0f;
    for (int k = 0; k < 10; ++k) a += sc[h][q][k] * vp[(long)(s * 10 + k) * 272 + c];
    o[(long)(s * 10 + q) * 272 + c] = a;
  }
}

// ---------------- row softmax for prop-1 scores ----------------
__global__ __launch_bounds__(256) void softmax_rows_kernel(float* P, int len) {
  float* p = P + (long)blockIdx.x * len;
  __shared__ float red[256];
  const int t = threadIdx.x;
  float mx = -3.0e38f;
  for (int i = t; i < len; i += 256) mx = fmaxf(mx, p[i]);
  red[t] = mx; __syncthreads();
  for (int s = 128; s > 0; s >>= 1) { if (t < s) red[t] = fmaxf(red[t], red[t + s]); __syncthreads(); }
  mx = red[0]; __syncthreads();
  float sum = 0.0f;
  for (int i = t; i < len; i += 256) { float e = expf(p[i] - mx); p[i] = e; sum += e; }
  red[t] = sum; __syncthreads();
  for (int s = 128; s > 0; s >>= 1) { if (t < s) red[t] += red[t + s]; __syncthreads(); }
  const float inv = 1.0f / red[0];
  __syncthreads();
  for (int i = t; i < len; i += 256) p[i] *= inv;
}

// ---------------- utility kernels ----------------
__global__ void cvt_bf16_kernel(const float* __restrict__ s, unsigned short* __restrict__ d, long n) {
  long i = (long)blockIdx.x * blockDim.x + threadIdx.x;
  const long st = (long)gridDim.x * blockDim.x;
  for (; i < n; i += st) d[i] = f32_to_bf16_rne(s[i]);
}
__global__ void copy_f32_kernel(const float* __restrict__ s, float* __restrict__ d, long n) {
  long i = (long)blockIdx.x * blockDim.x + threadIdx.x;
  const long st = (long)gridDim.x * blockDim.x;
  for (; i < n; i += st) d[i] = s[i];
}

// ---------------- host orchestration ----------------
extern "C" void kernel_launch(void* const* d_in, const int* in_sizes, int n_in,
                              void* d_out, int out_size, void* d_ws, size_t ws_size,
                              hipStream_t stream) {
  (void)in_sizes; (void)n_in; (void)out_size; (void)ws_size;
  const float* emb0 = (const float*)d_in[0];
  const float* edge0 = (const float*)d_in[1];
  const float* td0  = (const float*)d_in[2];
  const float* emb1 = (const float*)d_in[3];
  const float* edge1 = (const float*)d_in[4];
  const float* td1  = (const float*)d_in[5];
  const float* emb2 = (const float*)d_in[6];
  const float* Wq = (const float*)d_in[7];  const float* bq = (const float*)d_in[8];
  const float* Wk = (const float*)d_in[9];  const float* bk = (const float*)d_in[10];
  const float* Wv = (const float*)d_in[11]; const float* bv = (const float*)d_in[12];
  const float* Wo = (const float*)d_in[13]; const float* bo = (const float*)d_in[14];
  const float* W1 = (const float*)d_in[15]; const float* b1 = (const float*)d_in[16];
  const float* W2 = (const float*)d_in[17]; const float* b2 = (const float*)d_in[18];
  float* out = (float*)d_out;

  const int S = 2048, E = 272, KDT = 4440, FD = 172, FK = 444;
  const int M0 = S * 10;                 // 20480
  const float SCALE = 1.0f / sqrtf(136.0f);

  // ---- workspace layout ----
  unsigned char* ws = (unsigned char*)d_ws;
  size_t off = 0;
  auto take_bf = [&](size_t n) { unsigned short* p = (unsigned short*)(ws + off); off += n * 2; return p; };
  unsigned short* Wq_bf = take_bf((size_t)E * E);
  unsigned short* Wk_bf = take_bf((size_t)E * KDT);
  unsigned short* Wv_bf = take_bf((size_t)E * KDT);
  unsigned short* Wo_bf = take_bf((size_t)E * E);
  unsigned short* W1_bf = take_bf((size_t)FD * FK);
  unsigned short* W2_bf = take_bf((size_t)FD * FD);
  unsigned short* kp1_bf = take_bf((size_t)S * E);
  unsigned short* vp1_bf = take_bf((size_t)S * E);
  off = (off + 255) & ~(size_t)255;
  float* f32a = (float*)(ws + off);
  const long BUFE = (long)M0 * E;        // 5,570,560 floats
  float* buf0 = f32a;                    // qp0 -> a0 ; later P (spans buf0+buf1)
  float* buf1 = f32a + BUFE;             // kp0 -> h0
  float* buf2 = f32a + 2 * BUFE;         // vp0 -> qp1/kp1/vp1
  float* buf3 = f32a + 3 * BUFE;         // o0  -> o1/a1/h1
  float* qp1 = buf2;
  float* kp1 = buf2 + (long)S * E;
  float* vp1 = buf2 + 2L * S * E;
  float* P   = buf0;                     // 2*2048*2048 floats
  float* o1  = buf3;
  float* a1  = buf3 + (long)S * E;
  float* h1  = buf3 + 2L * S * E;

  auto cvt = [&](const float* s, unsigned short* d, long n) {
    int blocks = (int)((n + 1023) / 1024); if (blocks > 2048) blocks = 2048; if (blocks < 1) blocks = 1;
    cvt_bf16_kernel<<<blocks, 256, 0, stream>>>(s, d, n);
  };
  auto gemm = [&](GemmArgs g) {
    dim3 grid((g.N + TBN - 1) / TBN, (g.M + TBM - 1) / TBM);
    if (g.W2) gemm_bf16_wmma_kernel<true ><<<grid, 256, 0, stream>>>(g);
    else      gemm_bf16_wmma_kernel<false><<<grid, 256, 0, stream>>>(g);
  };
  auto mk = [&](const float* p0, long rs0, int w0,
                const float* p1, long rs1, int w1,
                const float* p2, long rs2, int w2,
                int period, int nGroup,
                const unsigned short* W, long ldW, int transB,
                const float* bias, float* o, long ldOut,
                int M, int N, int K, float alpha, int relu) {
    GemmArgs g{};
    g.s0 = {p0, rs0, w0, 0}; g.s1 = {p1, rs1, w1, 0}; g.s2 = {p2, rs2, w2, 0};
    g.period = period; g.nGroup = nGroup;
    g.W = W; g.W2 = nullptr; g.ldW = ldW; g.transB = transB;
    g.bias = bias; g.bias2 = nullptr; g.out = o; g.out2 = nullptr; g.ldOut = ldOut;
    g.M = M; g.N = N; g.K = K; g.alpha = alpha; g.relu = relu;
    return g;
  };

  // ---- weights to bf16 ----
  cvt(Wq, Wq_bf, (long)E * E);   cvt(Wk, Wk_bf, (long)E * KDT);
  cvt(Wv, Wv_bf, (long)E * KDT); cvt(Wo, Wo_bf, (long)E * E);
  cvt(W1, W1_bf, (long)FD * FK); cvt(W2, W2_bf, (long)FD * FD);

  // ================= propagation 0 =================
  // qp0 = (concat(emb1, td1) @ Wq^T + bq) * scale
  gemm(mk(emb1, 172, 172, td1, 100, 100, nullptr, 0, 0, E, 1,
          Wq_bf, E, 0, bq, buf0, E, M0, E, E, SCALE, 0));
  // kp0 & vp0 fused: shared gathered neigh0 tile, dual W (M=20480, K=4440)
  {
    GemmArgs g = mk(emb0, 172, 172, edge0, 172, 172, td0, 100, 100, 444, 10,
                    Wk_bf, KDT, 0, bk, buf1, E, M0, E, KDT, 1.0f, 0);
    g.W2 = Wv_bf; g.bias2 = bv; g.out2 = buf2;
    gemm(g);
  }
  // per-source 10x10 attention -> o0
  attn0_kernel<<<S, 256, 0, stream>>>(buf0, buf1, buf2, buf3);
  // a0 = o0 @ Wo^T + bo
  gemm(mk(buf3, E, E, nullptr, 0, 0, nullptr, 0, 0, E, 1,
          Wo_bf, E, 0, bo, buf0, E, M0, E, E, 1.0f, 0));
  // h0 = relu(concat(a0, emb1) @ W1^T + b1)
  gemm(mk(buf0, E, E, emb1, 172, 172, nullptr, 0, 0, FK, 1,
          W1_bf, FK, 0, b1, buf1, FD, M0, FD, FK, 1.0f, 1));
  // r0 = h0 @ W2^T + b2  -> d_out[0:20480*172]
  gemm(mk(buf1, FD, FD, nullptr, 0, 0, nullptr, 0, 0, FD, 1,
          W2_bf, FD, 0, b2, out, FD, M0, FD, FD, 1.0f, 0));

  // ================= propagation 1 =================
  // qp1 = (concat(emb2, zeros) @ Wq^T + bq) * scale
  gemm(mk(emb2, 172, 172, nullptr, 0, 100, nullptr, 0, 0, E, 1,
          Wq_bf, E, 0, bq, qp1, E, S, E, E, SCALE, 0));
  // kp1 & vp1 fused (M=2048, K=4440)
  {
    GemmArgs g = mk(emb1, 172, 172, edge1, 172, 172, td1, 100, 100, 444, 10,
                    Wk_bf, KDT, 0, bk, kp1, E, S, E, KDT, 1.0f, 0);
    g.W2 = Wv_bf; g.bias2 = bv; g.out2 = vp1;
    gemm(g);
  }
  cvt(kp1, kp1_bf, (long)S * E);
  cvt(vp1, vp1_bf, (long)S * E);
  // scores[h] = qp1_h @ kp1_h^T   (2048x2048, K=136)
  for (int h = 0; h < 2; ++h)
    gemm(mk(qp1 + h * 136, E, 136, nullptr, 0, 0, nullptr, 0, 0, 136, 1,
            kp1_bf + h * 136, E, 0, nullptr, P + (long)h * S * S, S,
            S, S, 136, 1.0f, 0));
  softmax_rows_kernel<<<2 * S, 256, 0, stream>>>(P, S);
  // o1[:, h*136:(h+1)*136] = P_h @ vp1_h   (W transposed view of vp1)
  for (int h = 0; h < 2; ++h)
    gemm(mk(P + (long)h * S * S, S, S, nullptr, 0, 0, nullptr, 0, 0, S, 1,
            vp1_bf + h * 136, E, 1, nullptr, o1 + h * 136, E,
            S, 136, S, 1.0f, 0));
  // a1 = o1 @ Wo^T + bo
  gemm(mk(o1, E, E, nullptr, 0, 0, nullptr, 0, 0, E, 1,
          Wo_bf, E, 0, bo, a1, E, S, E, E, 1.0f, 0));
  // h1 = relu(concat(a1, emb2) @ W1^T + b1)
  gemm(mk(a1, E, E, emb2, 172, 172, nullptr, 0, 0, FK, 1,
          W1_bf, FK, 0, b1, h1, FD, S, FD, FK, 1.0f, 1));
  // r1 = h1 @ W2^T + b2 -> d_out[3522560 : +352256]
  gemm(mk(h1, FD, FD, nullptr, 0, 0, nullptr, 0, 0, FD, 1,
          W2_bf, FD, 0, b2, out + 3522560L, FD, S, FD, FD, 1.0f, 0));

  // ---- passthrough outputs: edge1, td1 ----
  copy_f32_kernel<<<2048, 256, 0, stream>>>(edge1, out + 3874816L, 3522560L);
  copy_f32_kernel<<<2048, 256, 0, stream>>>(td1, out + 7397376L, 2048000L);
}
